// GraphNetwork_46720654246574
// MI455X (gfx1250) — compile-verified, matrix operand order checked by reading, not run on hardware
//
#include <hip/hip_runtime.h>
#include <stdint.h>
#include <stddef.h>

// ---------------- problem constants ----------------
#define N_NODES   4096
#define DEG       32
#define F_DIM     256
#define L_LAYERS  5
#define EPS_BN    1e-5f
#define NE_COUNT  (N_NODES * DEG)     // 131072 edges
#define POS_COUNT (N_NODES * F_DIM)   // 1048576 positions

// ---------------- CDNA5 WMMA types ----------------
typedef __attribute__((ext_vector_type(16))) __bf16 v16bf;
typedef __attribute__((ext_vector_type(8)))  float  v8f;

union FragBF { v16bf v; uint4 q[2]; };
union U8x16  { unsigned short s[8]; uint4 q; };

__device__ __forceinline__ unsigned short f2bf(float f) {
  unsigned u = __float_as_uint(f);
  u += 0x7FFFu + ((u >> 16) & 1u);          // round-to-nearest-even
  return (unsigned short)(u >> 16);
}
__device__ __forceinline__ float bf2f(unsigned short h) {
  return __uint_as_float(((unsigned)h) << 16);
}
__device__ __forceinline__ v8f vzero8() {
  v8f z;
#pragma unroll
  for (int i = 0; i < 8; ++i) z[i] = 0.f;
  return z;
}
__device__ __forceinline__ v8f wmma_bf16(v16bf a, v16bf b, v8f c) {
  return __builtin_amdgcn_wmma_f32_16x16x32_bf16(false, a, false, b, (short)0, c,
                                                 false, false);
}

// ---- CDNA5 async global->LDS DMA (ASYNCcnt-tracked, ISA 15.18.3 op 98) ----
// LDS destination address = low 32 bits of the generic shared pointer
// (ISA 10.2: LDS aperture addresses truncate to addr[31:0]).
__device__ __forceinline__ void async_ld_b128(void* lds_dst, const void* gsrc) {
  unsigned ldsa = (unsigned)(uintptr_t)lds_dst;
  unsigned long long ga = (unsigned long long)(uintptr_t)gsrc;
  asm volatile("global_load_async_to_lds_b128 %0, %1, off"
               :: "v"(ldsa), "v"(ga)
               : "memory");
}
__device__ __forceinline__ void wait_async0() {
  asm volatile("s_wait_asynccnt 0" ::: "memory");
}

// Load a 16x32 bf16 fragment (A from row-major rows, or B from column-major
// columns). ISA 7.12.2 16-bit A layout: lanes 0-15 hold K {0..7,16..23} of row
// `lane`; lanes 16-31 hold K {8..15,24..31}. Two 16B LDS loads per lane.
__device__ __forceinline__ v16bf ld_frag(const unsigned short* buf, int ld,
                                         int rbase, int kbase) {
  int lane = threadIdx.x & 31;
  int r  = rbase + (lane & 15);
  int k0 = kbase + ((lane & 16) >> 1);      // +8 for upper half-wave
  const unsigned short* p = buf + r * ld + k0;
  FragBF f;
  f.q[0] = *(const uint4*)(p);
  f.q[1] = *(const uint4*)(p + 16);
  return f.v;
}

// Store a 16x16 f32 accumulator tile as bf16 activations with bias + leaky
// ReLU. C/D layout: elem i of lane -> M = mbase + i + 8*(lane>=16),
// N = nbase + (lane & 15).
__device__ __forceinline__ void st_tile_lrelu(unsigned short* out, int ld,
                                              int mbase, int nbase, v8f acc,
                                              const float* bias) {
  int lane = threadIdx.x & 31;
  int n    = nbase + (lane & 15);
  int m0   = mbase + ((lane & 16) >> 1);
  float b  = bias[n];
#pragma unroll
  for (int i = 0; i < 8; ++i) {
    float x = acc[i] + b;
    x = (x > 0.f) ? x : 0.01f * x;
    out[(m0 + i) * ld + n] = f2bf(x);
  }
}

// ---------------- parameter plumbing ----------------
// JAX pytree flatten order (dict keys sorted) per layer, 34 leaves:
// eW1 eW2 eW3 eW4 eb1 eb2 eb3 eb4 ebn1{b,g,m,v} ebn2{..} ebn3{..}
// nW1 nW2 nW3 nb1 nb2 nb3 nbn1{b,g,m,v} nbn2{b,g,m,v}
struct LayerPtrs {
  const float *eW1, *eW2, *eW3, *eW4, *eb1, *eb2, *eb3, *eb4;
  const float *e1b, *e1g, *e1m, *e1v;
  const float *e2b, *e2g, *e2m, *e2v;
  const float *e3b, *e3g, *e3m, *e3v;
  const float *nW1, *nW2, *nW3, *nb1, *nb2, *nb3;
  const float *n1b, *n1g, *n1m, *n1v;
  const float *n2b, *n2g, *n2m, *n2v;
};
struct AllLayerPtrs { LayerPtrs l[L_LAYERS]; };

// Folded-weight block layout per layer (BN folded, bf16, column-major so a
// B-fragment column is contiguous in K):
#define WSL_EB1 0                         // u16[128][256]
#define WSL_EB2 (WSL_EB1 + 128*256*2)     // u16[64][128]
#define WSL_EB3 (WSL_EB2 + 64*128*2)      // u16[32][64]
#define WSL_NW2 (WSL_EB3 + 32*64*2)       // u16[32][64]
#define WSL_F   (WSL_NW2 + 32*64*2)       // float section
#define FB_EB1F 0     // 128
#define FB_EB2F 128   // 64
#define FB_EB3F 192   // 32
#define FB_EW4  224   // 32
#define FB_EB4  256   // 1
#define FB_NW1F 257   // 64
#define FB_NB1F 321   // 64
#define FB_NB2F 385   // 32
#define FB_NW3F 417   // 32
#define FB_NB3  449   // 1
#define WSL_BYTES (WSL_F + 512*4)         // 92160 B, 256B aligned

// Workspace layout
#define WS_NE   0
#define WS_AGG  (WS_NE + NE_COUNT * 4)
#define WS_NF16 (WS_AGG + POS_COUNT * 4)
#define WS_FOLD (WS_NF16 + POS_COUNT * 2)

// ---------------- kernels ----------------

// Fold BN into weights/biases; transpose weights to bf16 column-major.
__global__ void fold_params_kernel(AllLayerPtrs P, char* foldBase) {
  const LayerPtrs p = P.l[blockIdx.x];
  char* base = foldBase + (size_t)blockIdx.x * WSL_BYTES;
  unsigned short* eB1 = (unsigned short*)(base + WSL_EB1);
  unsigned short* eB2 = (unsigned short*)(base + WSL_EB2);
  unsigned short* eB3 = (unsigned short*)(base + WSL_EB3);
  unsigned short* nW2 = (unsigned short*)(base + WSL_NW2);
  float* fb = (float*)(base + WSL_F);
  int tid = threadIdx.x;

  for (int i = tid; i < 128 * 256; i += blockDim.x) {
    int n = i >> 8, k = i & 255;
    float s = p.e1g[n] * rsqrtf(p.e1v[n] + EPS_BN);
    eB1[i] = f2bf(p.eW1[k * 128 + n] * s);
  }
  for (int i = tid; i < 64 * 128; i += blockDim.x) {
    int n = i >> 7, k = i & 127;
    float s = p.e2g[n] * rsqrtf(p.e2v[n] + EPS_BN);
    eB2[i] = f2bf(p.eW2[k * 64 + n] * s);
  }
  for (int i = tid; i < 32 * 64; i += blockDim.x) {
    int n = i >> 6, k = i & 63;
    float s3 = p.e3g[n] * rsqrtf(p.e3v[n] + EPS_BN);
    eB3[i] = f2bf(p.eW3[k * 32 + n] * s3);
    float sn = p.n2g[n] * rsqrtf(p.n2v[n] + EPS_BN);
    nW2[i] = f2bf(p.nW2[k * 32 + n] * sn);
  }
  if (tid < 128) {
    float s = p.e1g[tid] * rsqrtf(p.e1v[tid] + EPS_BN);
    fb[FB_EB1F + tid] = (p.eb1[tid] - p.e1m[tid]) * s + p.e1b[tid];
  }
  if (tid < 64) {
    float s = p.e2g[tid] * rsqrtf(p.e2v[tid] + EPS_BN);
    fb[FB_EB2F + tid] = (p.eb2[tid] - p.e2m[tid]) * s + p.e2b[tid];
    float sn = p.n1g[tid] * rsqrtf(p.n1v[tid] + EPS_BN);
    fb[FB_NW1F + tid] = p.nW1[tid] * sn;
    fb[FB_NB1F + tid] = (p.nb1[tid] - p.n1m[tid]) * sn + p.n1b[tid];
  }
  if (tid < 32) {
    float s = p.e3g[tid] * rsqrtf(p.e3v[tid] + EPS_BN);
    fb[FB_EB3F + tid] = (p.eb3[tid] - p.e3m[tid]) * s + p.e3b[tid];
    fb[FB_EW4 + tid]  = p.eW4[tid];
    float sn = p.n2g[tid] * rsqrtf(p.n2v[tid] + EPS_BN);
    fb[FB_NB2F + tid] = (p.nb2[tid] - p.n2m[tid]) * sn + p.n2b[tid];
    fb[FB_NW3F + tid] = p.nW3[tid];
  }
  if (tid == 0) { fb[FB_EB4] = p.eb4[0]; fb[FB_NB3] = p.nb3[0]; }
}

__global__ void init_ne_kernel(const int* __restrict__ col,
                               const float* __restrict__ edge_feat,
                               float* __restrict__ ne) {
  int e = blockIdx.x * blockDim.x + threadIdx.x;
  if (e < NE_COUNT) ne[e] = edge_feat[col[e]];
}

// agg[n,f] = sum_d node_feat[col[n,d], f] * ne[n,d]  (block = node, tid = f)
__global__ void __launch_bounds__(256)
agg_kernel(const float* __restrict__ node_feat, const int* __restrict__ col,
           const float* __restrict__ ne, float* __restrict__ agg) {
  __shared__ int sc[DEG];
  __shared__ float sv[DEG];
  int n = blockIdx.x, tid = threadIdx.x;
  if (tid < DEG) { sc[tid] = col[n * DEG + tid]; sv[tid] = ne[n * DEG + tid]; }
  __syncthreads();
  float acc = 0.f;
#pragma unroll 8
  for (int d = 0; d < DEG; ++d) {
    if (d + 1 < DEG)
      __builtin_prefetch(node_feat + (size_t)sc[d + 1] * F_DIM + tid, 0, 0);
    acc = fmaf(node_feat[(size_t)sc[d] * F_DIM + tid], sv[d], acc);
  }
  agg[(size_t)n * F_DIM + tid] = acc;
}

// Node MLP: scalar x -> 64 (VALU, BN folded, lrelu) -> 32 (WMMA) -> sigmoid.
__global__ void __launch_bounds__(256)
node_kernel(const float* __restrict__ agg, const char* __restrict__ wl,
            unsigned short* __restrict__ nf16, float* __restrict__ nf32) {
  __shared__ unsigned short sA[256 * 64];   // bf16 h1 tile, A for WMMA
  __shared__ unsigned short sH2[256 * 32];  // bf16 h2 tile
  __shared__ unsigned short sW2[32 * 64];   // B col-major [32 cols][K=64]
  __shared__ float sW1[64], sB1[64], sB2[32], sW3[32], sB3s[1];
  const unsigned short* gW2 = (const unsigned short*)(wl + WSL_NW2);
  const float* fb = (const float*)(wl + WSL_F);
  int tid = threadIdx.x;

  // 4KB of weights: exactly one async b128 DMA per thread.
  async_ld_b128(sW2 + tid * 8, gW2 + tid * 8);
  if (tid < 64) { sW1[tid] = fb[FB_NW1F + tid]; sB1[tid] = fb[FB_NB1F + tid]; }
  if (tid < 32) { sB2[tid] = fb[FB_NB2F + tid]; sW3[tid] = fb[FB_NW3F + tid]; }
  if (tid == 0) sB3s[0] = fb[FB_NB3];

  int p = blockIdx.x * 256 + tid;
  float x = agg[p];
  wait_async0();
  __syncthreads();

#pragma unroll
  for (int j = 0; j < 64; j += 2) {         // pack bf16 pairs -> 32b stores
    float h0 = fmaf(x, sW1[j], sB1[j]);
    h0 = (h0 > 0.f) ? h0 : 0.01f * h0;
    float h1 = fmaf(x, sW1[j + 1], sB1[j + 1]);
    h1 = (h1 > 0.f) ? h1 : 0.01f * h1;
    unsigned pk = (unsigned)f2bf(h0) | ((unsigned)f2bf(h1) << 16);
    *(unsigned*)(sA + tid * 64 + j) = pk;
  }
  __syncthreads();

  // GEMM [256 x 64] @ [64 x 32]: 16 M-tiles x 2 N-tiles over 8 waves.
  int w = tid >> 5;
  int ni = w & 1;
  int mi0 = (w >> 1) * 4;
  v8f acc[4];
#pragma unroll
  for (int t = 0; t < 4; ++t) acc[t] = vzero8();
#pragma unroll
  for (int kc = 0; kc < 2; ++kc) {
    v16bf b = ld_frag(sW2, 64, ni * 16, kc * 32);
#pragma unroll
    for (int t = 0; t < 4; ++t) {
      v16bf a = ld_frag(sA, 64, (mi0 + t) * 16, kc * 32);
      acc[t] = wmma_bf16(a, b, acc[t]);
    }
  }
#pragma unroll
  for (int t = 0; t < 4; ++t)
    st_tile_lrelu(sH2, 32, (mi0 + t) * 16, ni * 16, acc[t], sB2);
  __syncthreads();

  float s = sB3s[0];
#pragma unroll
  for (int c = 0; c < 4; ++c) {             // b128 reads of h2
    U8x16 h;
    h.q = ((const uint4*)(sH2 + tid * 32))[c];
#pragma unroll
    for (int k = 0; k < 8; ++k) s = fmaf(bf2f(h.s[k]), sW3[c * 8 + k], s);
  }
  float sg = 1.f / (1.f + __expf(-s));
  nf16[p] = f2bf(sg);
  if (nf32) nf32[p] = sg;
}

// Edge MLP: persistent kernel; weights staged in LDS once via async DMA;
// 64-edge tiles through three WMMA GEMM stages.
#define ET 64
constexpr unsigned EDGE_LDS_BYTES =
    (ET*256 + 128*256 + ET*128 + 64*128 + ET*64 + 32*64 + ET*32) * 2 + 260 * 4;

__global__ void __launch_bounds__(256)
edge_kernel(const unsigned short* __restrict__ nf16,
            const int* __restrict__ col, float* __restrict__ ne,
            const char* __restrict__ wl, int ntiles) {
  extern __shared__ char smem[];
  unsigned short* sDiff = (unsigned short*)smem;      // [64][256]
  unsigned short* sB1   = sDiff + ET * 256;           // [128][256] col-major
  unsigned short* sH1   = sB1 + 128 * 256;            // [64][128]
  unsigned short* sB2   = sH1 + ET * 128;             // [64][128] col-major
  unsigned short* sH2   = sB2 + 64 * 128;             // [64][64]
  unsigned short* sB3   = sH2 + ET * 64;              // [32][64] col-major
  unsigned short* sH3   = sB3 + 32 * 64;              // [64][32]
  float* sF  = (float*)(sH3 + ET * 32);
  float* sb1 = sF;            // 128
  float* sb2 = sF + 128;      // 64
  float* sb3 = sF + 192;      // 32
  float* sw4 = sF + 224;      // 32; sF[256] = eb4

  const unsigned short* gB1 = (const unsigned short*)(wl + WSL_EB1);
  const unsigned short* gB2 = (const unsigned short*)(wl + WSL_EB2);
  const unsigned short* gB3 = (const unsigned short*)(wl + WSL_EB3);
  const float* fb = (const float*)(wl + WSL_F);
  int tid = threadIdx.x;
  // 84KB of folded weights: straight global->LDS DMA, no VGPR round-trip.
  for (int i = tid; i < (128 * 256) / 8; i += 256)
    async_ld_b128(sB1 + i * 8, gB1 + i * 8);
  for (int i = tid; i < (64 * 128) / 8; i += 256)
    async_ld_b128(sB2 + i * 8, gB2 + i * 8);
  for (int i = tid; i < (32 * 64) / 8; i += 256)
    async_ld_b128(sB3 + i * 8, gB3 + i * 8);
  if (tid < 128) sb1[tid] = fb[FB_EB1F + tid];
  if (tid < 64)  sb2[tid] = fb[FB_EB2F + tid];
  if (tid < 32)  { sb3[tid] = fb[FB_EB3F + tid]; sw4[tid] = fb[FB_EW4 + tid]; }
  if (tid == 0)  sF[256] = fb[FB_EB4];
  wait_async0();
  __syncthreads();

  int w = tid >> 5, lane = tid & 31;
  for (int tile = blockIdx.x; tile < ntiles; tile += gridDim.x) {
    int e0 = tile * ET;
    // Build |nf_i - nf_j| tile: wave w handles 8 edges; lane covers 8 feats.
#pragma unroll
    for (int s = 0; s < 8; ++s) {
      int e = e0 + w * 8 + s;
      int i = e >> 5;                 // row node (DEG = 32)
      int j = col[e];
      U8x16 ua, ub, ud;
      ua.q = ((const uint4*)(nf16 + (size_t)i * F_DIM))[lane];
      ub.q = ((const uint4*)(nf16 + (size_t)j * F_DIM))[lane];
#pragma unroll
      for (int k = 0; k < 8; ++k)
        ud.s[k] = f2bf(fabsf(bf2f(ua.s[k]) - bf2f(ub.s[k])));
      ((uint4*)(sDiff + (w * 8 + s) * F_DIM))[lane] = ud.q;
    }
    __syncthreads();

    // GEMM1: [64 x 256] @ [256 x 128]; wave w owns N-strip ni = w.
    {
      v8f acc[4];
#pragma unroll
      for (int t = 0; t < 4; ++t) acc[t] = vzero8();
#pragma unroll
      for (int kc = 0; kc < 8; ++kc) {
        v16bf b = ld_frag(sB1, 256, w * 16, kc * 32);
#pragma unroll
        for (int t = 0; t < 4; ++t) {
          v16bf a = ld_frag(sDiff, 256, t * 16, kc * 32);
          acc[t] = wmma_bf16(a, b, acc[t]);
        }
      }
#pragma unroll
      for (int t = 0; t < 4; ++t)
        st_tile_lrelu(sH1, 128, t * 16, w * 16, acc[t], sb1);
    }
    __syncthreads();

    // GEMM2: [64 x 128] @ [128 x 64]; 4x4 tiles over 8 waves (2 each).
    {
      int ni = w & 3, mi0 = (w >> 2) * 2;
      v8f acc[2];
      acc[0] = vzero8(); acc[1] = vzero8();
#pragma unroll
      for (int kc = 0; kc < 4; ++kc) {
        v16bf b = ld_frag(sB2, 128, ni * 16, kc * 32);
#pragma unroll
        for (int t = 0; t < 2; ++t) {
          v16bf a = ld_frag(sH1, 128, (mi0 + t) * 16, kc * 32);
          acc[t] = wmma_bf16(a, b, acc[t]);
        }
      }
#pragma unroll
      for (int t = 0; t < 2; ++t)
        st_tile_lrelu(sH2, 64, (mi0 + t) * 16, ni * 16, acc[t], sb2);
    }
    __syncthreads();

    // GEMM3: [64 x 64] @ [64 x 32]; 4x2 tiles over 8 waves (1 each).
    {
      int ni = w >> 2, mi = w & 3;
      v8f acc = vzero8();
#pragma unroll
      for (int kc = 0; kc < 2; ++kc) {
        v16bf b = ld_frag(sB3, 64, ni * 16, kc * 32);
        v16bf a = ld_frag(sH2, 64, mi * 16, kc * 32);
        acc = wmma_bf16(a, b, acc);
      }
      st_tile_lrelu(sH3, 32, mi * 16, ni * 16, acc, sb3);
    }
    __syncthreads();

    // Final 32 -> 1 dot + sigmoid; ne += upd.
    if (tid < ET) {
      float s = sF[256];
#pragma unroll
      for (int c = 0; c < 4; ++c) {
        U8x16 h;
        h.q = ((const uint4*)(sH3 + tid * 32))[c];
#pragma unroll
        for (int k = 0; k < 8; ++k)
          s = fmaf(bf2f(h.s[k]), sw4[c * 8 + k], s);
      }
      float u = 1.f / (1.f + __expf(-s));
      ne[e0 + tid] += u;
    }
    __syncthreads();
  }
}

// logits[n, col[n,d]] += ne[n,d]; zero-fill row then 32 atomics.
__global__ void __launch_bounds__(256)
scatter_logits_kernel(const float* __restrict__ ne, const int* __restrict__ col,
                      float* __restrict__ out) {
  __shared__ int sc[DEG];
  __shared__ float sv[DEG];
  int n = blockIdx.x, tid = threadIdx.x;
  if (tid < DEG) { sc[tid] = col[n * DEG + tid]; sv[tid] = ne[n * DEG + tid]; }
  float* row = out + (size_t)n * N_NODES;
  float4 z = {0.f, 0.f, 0.f, 0.f};
  for (int i = tid; i < N_NODES / 4; i += 256) ((float4*)row)[i] = z;
  __threadfence();
  __syncthreads();
  if (tid < DEG) atomicAdd(row + sc[tid], sv[tid]);
}

// ---------------- launcher ----------------
extern "C" void kernel_launch(void* const* d_in, const int* in_sizes, int n_in,
                              void* d_out, int out_size, void* d_ws,
                              size_t ws_size, hipStream_t stream) {
  (void)in_sizes; (void)n_in; (void)out_size; (void)ws_size;
  const float* node_feat  = (const float*)d_in[0];
  const int*   edge_index = (const int*)d_in[1];       // [2][N*D], row then col
  const int*   col        = edge_index + NE_COUNT;
  const float* edge_feat  = (const float*)d_in[2];

  AllLayerPtrs P;
  for (int l = 0; l < L_LAYERS; ++l) {
    const float* const* q = (const float* const*)(d_in + 3 + l * 34);
    LayerPtrs& p = P.l[l];
    p.eW1 = q[0];  p.eW2 = q[1];  p.eW3 = q[2];  p.eW4 = q[3];
    p.eb1 = q[4];  p.eb2 = q[5];  p.eb3 = q[6];  p.eb4 = q[7];
    p.e1b = q[8];  p.e1g = q[9];  p.e1m = q[10]; p.e1v = q[11];
    p.e2b = q[12]; p.e2g = q[13]; p.e2m = q[14]; p.e2v = q[15];
    p.e3b = q[16]; p.e3g = q[17]; p.e3m = q[18]; p.e3v = q[19];
    p.nW1 = q[20]; p.nW2 = q[21]; p.nW3 = q[22];
    p.nb1 = q[23]; p.nb2 = q[24]; p.nb3 = q[25];
    p.n1b = q[26]; p.n1g = q[27]; p.n1m = q[28]; p.n1v = q[29];
    p.n2b = q[30]; p.n2g = q[31]; p.n2m = q[32]; p.n2v = q[33];
  }

  char* ws = (char*)d_ws;
  float* ne            = (float*)(ws + WS_NE);
  float* agg           = (float*)(ws + WS_AGG);
  unsigned short* nf16 = (unsigned short*)(ws + WS_NF16);
  char* wfold          = ws + WS_FOLD;

  fold_params_kernel<<<L_LAYERS, 256, 0, stream>>>(P, wfold);
  init_ne_kernel<<<(NE_COUNT + 255) / 256, 256, 0, stream>>>(col, edge_feat, ne);

  float* nf_out = (float*)d_out;
  float* logits = (float*)d_out + POS_COUNT;
  const int ntiles = NE_COUNT / ET;   // 2048

  for (int l = 0; l < L_LAYERS; ++l) {
    const char* wl = wfold + (size_t)l * WSL_BYTES;
    agg_kernel<<<N_NODES, 256, 0, stream>>>(node_feat, col, ne, agg);
    node_kernel<<<POS_COUNT / 256, 256, 0, stream>>>(
        agg, wl, nf16, (l == L_LAYERS - 1) ? nf_out : nullptr);
    edge_kernel<<<1024, 256, EDGE_LDS_BYTES, stream>>>(nf16, col, ne, wl, ntiles);
    scatter_logits_kernel<<<N_NODES, 256, 0, stream>>>(
        ne, col, logits + (size_t)l * N_NODES * N_NODES);
  }
}
